// Net_74440373174447
// MI455X (gfx1250) — matrix operand order repaired and checked by reading, not compiled, not run on hardware
//
#include <hip/hip_runtime.h>
#include <cmath>

typedef __attribute__((ext_vector_type(16))) _Float16 v16h;
typedef __attribute__((ext_vector_type(8)))  _Float16 v8h;
typedef __attribute__((ext_vector_type(8)))  float    v8f;

#define EPSF 1e-15f
__device__ __forceinline__ float clipn(float x) { return fmaxf(x, EPSF); }

// ---------------------------------------------------------------------------
// Generic hyperbolic stage: tangent-in (NHWC), im2col(+beta) -> expmap0 ->
// PoincareFC -> logmap0 -> tangent-out (NHWC).  GEMM via v_wmma_f32_16x16x32_f16.
// Block = 256 threads (8 waves); each wave owns a 16-row M tile; 128 rows/block.
// KSZ/CIN are compile-time so the im2col index math is div-free.
// ---------------------------------------------------------------------------
template <int NFRAG, int KSZ, int CIN>
__global__ __launch_bounds__(256) void hstage(
    const float* __restrict__ in, const _Float16* __restrict__ packB,
    const float* __restrict__ tab, float* __restrict__ out,
    int Hin, int Win, int Hout, int Wout, int Cout,
    int K, int Kchunks, int Mtotal, float sc, float beta)
{
    __shared__ alignas(16) _Float16 Atile[128][32];
    __shared__ float rowp[128][2];

    const int tid  = threadIdx.x;
    const int wave = tid >> 5;
    const int lane = tid & 31;

    // im2col fill role: two threads per row
    const int frow = tid >> 1;
    const int fkb  = (tid & 1) * 16;
    const int fgm  = blockIdx.x * 128 + frow;
    const bool fvalid = (fgm < Mtotal);
    const float* rowbase = in;
    if (fvalid) {
        int fox = fgm % Wout;
        int t   = fgm / Wout;
        int foy = t % Hout;
        int fb  = t / Hout;
        rowbase = in + (((size_t)fb * Hin + foy) * Win + fox) * CIN;
    }

    v8f acc[NFRAG] = {};
    const v16h* bp = (const v16h*)packB;
    float ssacc = 0.0f;

    for (int kc = 0; kc < Kchunks; ++kc) {
        // ---- im2col chunk into LDS (f16, beta-scaled), accumulate row sumsq
        #pragma unroll
        for (int e = 0; e < 16; ++e) {
            int kk = kc * 32 + fkb + e;
            float v = 0.0f;
            if (fvalid && kk < K) {
                int kx = kk % KSZ;                      // compile-time divisors
                int t  = kk / KSZ;
                int ky = t % KSZ;
                int c  = t / KSZ;                       // K order = ((c*k)+ky)*k+kx
                v = rowbase[((size_t)ky * Win + kx) * CIN + c] * beta;
            }
            ssacc += v * v;
            Atile[frow][fkb + e] = (_Float16)v;
        }
        __syncthreads();

        // ---- A fragment per documented 16-bit 16x32 layout
        const int arow = (wave << 4) + (lane & 15);
        const int hi   = lane >> 4;                     // lanes 16-31 -> K+8 groups
        const v8h* rp  = (const v8h*)(&Atile[arow][0]);
        v8h alo = rp[hi];                               // K 0..7   / 8..15
        v8h ahi = rp[hi + 2];                           // K 16..23 / 24..31
        v16h a;
        #pragma unroll
        for (int i = 0; i < 8; ++i) { a[i] = alo[i]; a[8 + i] = ahi[i]; }

        if (kc + 1 < Kchunks)
            __builtin_prefetch(bp + (size_t)((kc + 1) * NFRAG) * 32 + lane, 0, 0);

        #pragma unroll
        for (int f = 0; f < NFRAG; ++f) {
            v16h b = bp[(size_t)(kc * NFRAG + f) * 32 + lane];
            acc[f] = __builtin_amdgcn_wmma_f32_16x16x32_f16(
                false, a, false, b, (short)0, acc[f], false, false);
        }
        __syncthreads();
    }

    rowp[frow][tid & 1] = ssacc;
    __syncthreads();

    // ---- epilogue: lane holds col (lane&15) of rows rbase..rbase+7
    const int col0  = lane & 15;
    const int rbase = (wave << 4) + ((lane >> 4) << 3);

    float lam8[8], sexp8[8];
    #pragma unroll
    for (int i = 0; i < 8; ++i) {
        float np = clipn(sqrtf(rowp[rbase + i][0] + rowp[rbase + i][1]));
        float aa = sc * np;
        float ch = coshf(aa);
        lam8[i]  = 2.0f * ch * ch;                      // = 2/(1 - c|x|^2)
        sexp8[i] = tanhf(aa) / aa;                      // expmap0 scale
    }

    float w[NFRAG][8];
    float rs[8];
    #pragma unroll
    for (int i = 0; i < 8; ++i) rs[i] = 0.0f;

    #pragma unroll
    for (int f = 0; f < NFRAG; ++f) {
        int j = f * 16 + col0;
        float zn  = tab[j * 3 + 0];
        float chr = tab[j * 3 + 1];                     // cosh(2 sc r)
        float shr = tab[j * 3 + 2];                     // sinh(2 sc r)
        #pragma unroll
        for (int i = 0; i < 8; ++i) {
            float xz  = sexp8[i] * acc[f][i];
            float arg = sc * lam8[i] * xz / zn * chr - (lam8[i] - 1.0f) * shr;
            float vv  = (2.0f * zn / sc) * asinhf(arg);
            float ww  = sinhf(sc * vv);                 // ww = sc * y_j
            w[f][i] = ww;
            rs[i] += ww * ww;
        }
    }
    // sum |w|^2 over the 16 lanes holding this row's columns
    #pragma unroll
    for (int m = 1; m < 16; m <<= 1) {
        #pragma unroll
        for (int i = 0; i < 8; ++i) rs[i] += __shfl_xor(rs[i], m, 16);
    }

    #pragma unroll
    for (int i = 0; i < 8; ++i) {
        int gm = blockIdx.x * 128 + rbase + i;
        if (gm >= Mtotal) continue;
        float t     = sqrtf(rs[i]);                     // = sc*|y|
        float tansc = asinhf(t) / (2.0f * sc * clipn(t));  // proj+logmap0 fused
        #pragma unroll
        for (int f = 0; f < NFRAG; ++f) {
            int j = f * 16 + col0;
            if (j < Cout) out[(size_t)gm * Cout + j] = tansc * w[f][i];
        }
    }
}

// ---------------------------------------------------------------------------
// Weight packer: f16 B fragments in per-lane WMMA order + per-column tables
// (zn, cosh(2 sc r), sinh(2 sc r)); padded cols get (1,1,0) -> exact zeros.
// ---------------------------------------------------------------------------
__global__ void pack_weights(const float* __restrict__ z, const float* __restrict__ r,
                             _Float16* __restrict__ pack, float* __restrict__ tab,
                             int K, int Cout, int Kchunks, int Nfrag, float sc)
{
    int total = Kchunks * Nfrag * 512;
    for (int i = blockIdx.x * blockDim.x + threadIdx.x; i < total;
         i += gridDim.x * blockDim.x) {
        int e    = i & 15;
        int lane = (i >> 4) & 31;
        int f    = (i >> 9) % Nfrag;
        int kc   = i / (Nfrag * 512);
        int n    = f * 16 + (lane & 15);
        int kk   = kc * 32 + ((e >> 3) << 4) + ((lane >> 4) << 3) + (e & 7); // B 32x16 lane layout (table order)
        float v  = (kk < K && n < Cout) ? z[(size_t)kk * Cout + n] : 0.0f;
        pack[i]  = (_Float16)v;
    }
    int j = blockIdx.x * blockDim.x + threadIdx.x;
    int Npad = Nfrag * 16;
    if (j < Npad) {
        float zn = 1.0f, ch = 1.0f, sh = 0.0f;
        if (j < Cout) {
            float s = 0.0f;
            for (int kk = 0; kk < K; ++kk) { float v = z[(size_t)kk * Cout + j]; s += v * v; }
            zn = fmaxf(sqrtf(s), EPSF);
            float rr = r[j];
            ch = coshf(2.0f * sc * rr);
            sh = sinhf(2.0f * sc * rr);
        }
        tab[j * 3 + 0] = zn; tab[j * 3 + 1] = ch; tab[j * 3 + 2] = sh;
    }
}

// ---- tangent-space maxpool + relu (NHWC) ----------------------------------
__global__ void hmaxpool_relu(const float* __restrict__ in, float* __restrict__ out,
                              int H, int W, int C, int Ho, int Wo, int pk, int ps, int n)
{
    int i = blockIdx.x * blockDim.x + threadIdx.x;
    if (i >= n) return;
    int c = i % C; int t = i / C;
    int wo = t % Wo; t /= Wo;
    int ho = t % Ho; int b = t / Ho;
    float m = -INFINITY;
    for (int ky = 0; ky < pk; ++ky)
        for (int kx = 0; kx < pk; ++kx)
            m = fmaxf(m, in[(((size_t)b * H + ho * ps + ky) * W + wo * ps + kx) * C + c]);
    out[i] = fmaxf(m, 0.0f);
}

// ---- tangent-space relu then global average (conv3 -> hrelu -> havgpool) ---
__global__ void hrelu_avg(const float* __restrict__ in, float* __restrict__ out,
                          int HW, int C, int n)
{
    int i = blockIdx.x * blockDim.x + threadIdx.x;
    if (i >= n) return;
    int c = i % C; int b = i / C;
    float s = 0.0f;
    for (int p = 0; p < HW; ++p) s += fmaxf(in[((size_t)b * HW + p) * C + c], 0.0f);
    out[i] = s / (float)HW;
}

__global__ void relu_vec(const float* __restrict__ in, float* __restrict__ out, int n)
{
    int i = blockIdx.x * blockDim.x + threadIdx.x;
    if (i < n) out[i] = fmaxf(in[i], 0.0f);
}

// ---------------------------------------------------------------------------
static double lb_half(double a) { return lgamma(a * 0.5) + lgamma(0.5) - lgamma(a * 0.5 + 0.5); }
static float beta_ratio_h(int n, int m) { return (float)exp(lb_half((double)n) - lb_half((double)m)); }

extern "C" void kernel_launch(void* const* d_in, const int* in_sizes, int n_in,
                              void* d_out, int out_size, void* d_ws, size_t ws_size,
                              hipStream_t stream)
{
    (void)in_sizes; (void)n_in; (void)out_size; (void)ws_size;
    const float* x   = (const float*)d_in[0];
    const float* z1  = (const float*)d_in[1];  const float* b1  = (const float*)d_in[2];
    const float* z2  = (const float*)d_in[3];  const float* b2  = (const float*)d_in[4];
    const float* z3  = (const float*)d_in[5];  const float* b3  = (const float*)d_in[6];
    const float* zf1 = (const float*)d_in[7];  const float* bf1 = (const float*)d_in[8];
    const float* zf2 = (const float*)d_in[9];  const float* bf2 = (const float*)d_in[10];
    float* outp = (float*)d_out;
    char*  ws   = (char*)d_ws;

    const float sc = sqrtf(0.1f);                 // C0 == C1 == 0.1
    const float beta1 = beta_ratio_h(25, 1);
    const float beta2 = beta_ratio_h(384, 24);
    const float beta3 = beta_ratio_h(324, 36);

    // ---- workspace layout (bytes) ----
    float*    tab1 = (float*)(ws + 0);            // 32*3
    float*    tab2 = (float*)(ws + 384);          // 48*3
    float*    tab3 = (float*)(ws + 960);          // 48*3
    float*    tab4 = (float*)(ws + 1536);         // 64*3
    float*    tab5 = (float*)(ws + 2304);         // 16*3
    _Float16* pk1  = (_Float16*)(ws + 4096);      // 1*2*512  halves
    _Float16* pk2  = (_Float16*)(ws + 6144);      // 12*3*512
    _Float16* pk3  = (_Float16*)(ws + 43008);     // 11*3*512
    _Float16* pk4  = (_Float16*)(ws + 76800);     // 2*4*512
    _Float16* pk5  = (_Float16*)(ws + 84992);     // 2*1*512
    float*    t3   = (float*)(ws + 98304);        // 128*48
    float*    fc1b = (float*)(ws + 122880);       // 128*60
    float*    t4   = (float*)(ws + 153600);       // 128*60
    float*    bufA = (float*)(ws + (1u << 20));   // big conv buffer (reused)
    float*    t1   = (float*)(ws + (1u << 20) + 188964864u);  // 128*41*41*24
    float*    t2   = (float*)(ws + (1u << 20) + 188964864u + 20654592u); // 128*19*19*36

    // ---- pack weights + column tables ----
    pack_weights<<<5,  256, 0, stream>>>(z1,  b1,  pk1, tab1, 25,  24, 1,  2, sc);
    pack_weights<<<72, 256, 0, stream>>>(z2,  b2,  pk2, tab2, 384, 36, 12, 3, sc);
    pack_weights<<<66, 256, 0, stream>>>(z3,  b3,  pk3, tab3, 324, 48, 11, 3, sc);
    pack_weights<<<16, 256, 0, stream>>>(zf1, bf1, pk4, tab4, 48,  60, 2,  4, sc);
    pack_weights<<<4,  256, 0, stream>>>(zf2, bf2, pk5, tab5, 60,  10, 2,  1, sc);

    // ---- stage 1: conv 5x5, 1->24, 128 -> 124 ----
    hstage<2, 5, 1><<<15376, 256, 0, stream>>>(x, pk1, tab1, bufA,
        128, 128, 124, 124, 24, 25, 1, 128 * 124 * 124, sc, beta1);
    {   int n = 128 * 41 * 41 * 24;
        hmaxpool_relu<<<(n + 255) / 256, 256, 0, stream>>>(bufA, t1, 124, 124, 24, 41, 41, 3, 3, n); }

    // ---- stage 2: conv 4x4, 24->36, 41 -> 38 ----
    hstage<3, 4, 24><<<1444, 256, 0, stream>>>(t1, pk2, tab2, bufA,
        41, 41, 38, 38, 36, 384, 12, 128 * 38 * 38, sc, beta2);
    {   int n = 128 * 19 * 19 * 36;
        hmaxpool_relu<<<(n + 255) / 256, 256, 0, stream>>>(bufA, t2, 38, 38, 36, 19, 19, 2, 2, n); }

    // ---- stage 3: conv 3x3, 36->48, 19 -> 17 ----
    hstage<3, 3, 36><<<289, 256, 0, stream>>>(t2, pk3, tab3, bufA,
        19, 19, 17, 17, 48, 324, 11, 128 * 17 * 17, sc, beta3);
    {   int n = 128 * 48;
        hrelu_avg<<<(n + 255) / 256, 256, 0, stream>>>(bufA, t3, 289, 48, n); }

    // ---- FC1 (48->60) as k=1 stage on C1 manifold ----
    hstage<4, 1, 48><<<1, 256, 0, stream>>>(t3, pk4, tab4, fc1b,
        1, 1, 1, 1, 60, 48, 2, 128, sc, 1.0f);
    relu_vec<<<30, 256, 0, stream>>>(fc1b, t4, 128 * 60);

    // ---- FC2 (60->10): output tangent == final logits ----
    hstage<1, 1, 60><<<1, 256, 0, stream>>>(t4, pk5, tab5, outp,
        1, 1, 1, 1, 10, 60, 2, 128, sc, 1.0f);
}
